// STAG_78786880077949
// MI455X (gfx1250) — matrix operand order repaired and checked by reading, not compiled, not run on hardware
//
#include <hip/hip_runtime.h>

#define NN 16384
#define NE 131072
#define NG 64
#define EPSV 1e-5f
#define SLOPE 0.01f

typedef __attribute__((ext_vector_type(16))) _Float16 v16h;
typedef __attribute__((ext_vector_type(8)))  float    v8f;

__device__ __forceinline__ float lrelu(float x) { return x > 0.f ? x : SLOPE * x; }
__device__ __forceinline__ unsigned fenc(float x) {
    unsigned u = __float_as_uint(x);
    return (u & 0x80000000u) ? ~u : (u | 0x80000000u);
}
__device__ __forceinline__ float fdec(unsigned u) {
    return __uint_as_float((u & 0x80000000u) ? (u & 0x7fffffffu) : ~u);
}

// ---------------- fills ----------------
__global__ void k_fill_f32(float* p, float v, int n) {
    int i = blockIdx.x * blockDim.x + threadIdx.x;
    if (i < n) p[i] = v;
}
__global__ void k_fill_u32(unsigned* p, unsigned v, int n) {
    int i = blockIdx.x * blockDim.x + threadIdx.x;
    if (i < n) p[i] = v;
}

// ---------------- node count per graph ----------------
__global__ void k_count(const int* __restrict__ batch, float* __restrict__ cnt) {
    int i = blockIdx.x * blockDim.x + threadIdx.x;
    if (i < NN) atomicAdd(&cnt[batch[i]], 1.0f);
}

// ---------------- generic linear: out[n,32] = in[n,fi] @ W[fi,32] + b ----------------
__global__ void k_lin(const float* __restrict__ in, const float* __restrict__ W,
                      const float* __restrict__ b, float* __restrict__ out,
                      int n, int fi) {
    int t = blockIdx.x * blockDim.x + threadIdx.x;
    int node = t >> 5, o = t & 31;
    if (node >= n) return;
    float s = b[o];
    const float* r = in + (size_t)node * fi;
    for (int c = 0; c < fi; ++c) s = fmaf(r[c], W[c * 32 + o], s);
    out[(size_t)node * 32 + o] = s;
}

// ---------------- graph layernorm (PyG mode='graph') ----------------
__global__ void k_gln_reduce(const float* __restrict__ v, const int* __restrict__ batch,
                             float* __restrict__ gsum, float* __restrict__ gsq) {
    int i = blockIdx.x * blockDim.x + threadIdx.x;
    if (i >= NN) return;
    const float* r = v + (size_t)i * 32;
    float s = 0.f, s2 = 0.f;
#pragma unroll
    for (int f = 0; f < 32; ++f) { float x = r[f]; s += x; s2 = fmaf(x, x, s2); }
    int g = batch[i];
    atomicAdd(&gsum[g], s);
    atomicAdd(&gsq[g], s2);
}
__global__ void k_gln_final(const float* __restrict__ gsum, const float* __restrict__ gsq,
                            const float* __restrict__ cnt, float* __restrict__ gmean,
                            float* __restrict__ ginv) {
    int g = threadIdx.x;
    if (g >= NG) return;
    float norm = fmaxf(cnt[g], 1.f) * 32.f;
    float m = gsum[g] / norm;
    float var = gsq[g] / norm - m * m;
    gmean[g] = m;
    ginv[g] = rsqrtf(var + EPSV);
}
__global__ void k_gln_apply(const float* __restrict__ in, float* __restrict__ out,
                            const int* __restrict__ batch, const float* __restrict__ gmean,
                            const float* __restrict__ ginv, const float* __restrict__ w,
                            const float* __restrict__ b, int leaky) {
    int t = blockIdx.x * blockDim.x + threadIdx.x;
    int node = t >> 5, f = t & 31;
    if (node >= NN) return;
    int g = batch[node];
    float x = (in[(size_t)node * 32 + f] - gmean[g]) * ginv[g] * w[f] + b[f];
    if (leaky) x = lrelu(x);
    out[(size_t)node * 32 + f] = x;
}

// ---------------- edge features: concat(h[row],h[col],eattr)@We1 -> LN16 -> lrelu -> @We2 ----------------
__global__ void __launch_bounds__(256) k_edge_feat(
    const float* __restrict__ h, const int* __restrict__ ei, const float* __restrict__ eattr,
    const float* __restrict__ We1, const float* __restrict__ be1,
    const float* __restrict__ enw, const float* __restrict__ enb,
    const float* __restrict__ We2, const float* __restrict__ be2,
    float* __restrict__ ea) {
    __shared__ float sW1[83 * 16];
    __shared__ float sW2[16 * 16];
    __shared__ float sb1[16], sw[16], sbb[16], sb2[16];
    for (int i = threadIdx.x; i < 83 * 16; i += 256) sW1[i] = We1[i];
    for (int i = threadIdx.x; i < 16 * 16; i += 256) sW2[i] = We2[i];
    if (threadIdx.x < 16) {
        sb1[threadIdx.x] = be1[threadIdx.x];
        sw[threadIdx.x] = enw[threadIdx.x];
        sbb[threadIdx.x] = enb[threadIdx.x];
        sb2[threadIdx.x] = be2[threadIdx.x];
    }
    __syncthreads();
    int e = blockIdx.x * 256 + threadIdx.x;
    if (e >= NE) return;
    int r = ei[e], c = ei[NE + e];
    float acc[16];
#pragma unroll
    for (int o = 0; o < 16; ++o) acc[o] = sb1[o];
    const float* hr = h + (size_t)r * 32;
    const float* hc = h + (size_t)c * 32;
    for (int f = 0; f < 32; ++f) {
        float x = hr[f];
#pragma unroll
        for (int o = 0; o < 16; ++o) acc[o] = fmaf(x, sW1[f * 16 + o], acc[o]);
    }
    for (int f = 0; f < 32; ++f) {
        float x = hc[f];
#pragma unroll
        for (int o = 0; o < 16; ++o) acc[o] = fmaf(x, sW1[(32 + f) * 16 + o], acc[o]);
    }
    const float* ep = eattr + (size_t)e * 19;
    for (int f = 0; f < 19; ++f) {
        float x = ep[f];
#pragma unroll
        for (int o = 0; o < 16; ++o) acc[o] = fmaf(x, sW1[(64 + f) * 16 + o], acc[o]);
    }
    float m = 0.f;
#pragma unroll
    for (int o = 0; o < 16; ++o) m += acc[o];
    m *= (1.f / 16.f);
    float var = 0.f;
#pragma unroll
    for (int o = 0; o < 16; ++o) { float d = acc[o] - m; var = fmaf(d, d, var); }
    var *= (1.f / 16.f);
    float inv = rsqrtf(var + EPSV);
    float hmid[16];
#pragma unroll
    for (int o = 0; o < 16; ++o) hmid[o] = lrelu((acc[o] - m) * inv * sw[o] + sbb[o]);
    float* op = ea + (size_t)e * 16;
#pragma unroll
    for (int o = 0; o < 16; ++o) {
        float s = sb2[o];
#pragma unroll
        for (int c2 = 0; c2 < 16; ++c2) s = fmaf(hmid[c2], sW2[c2 * 16 + o], s);
        op[o] = s;
    }
}

// ---------------- attention logits + segment max ----------------
__global__ void k_logits(const float* __restrict__ q, const float* __restrict__ k,
                         const int* __restrict__ ei, float* __restrict__ logit,
                         unsigned* __restrict__ mbuf) {
    int e = blockIdx.x * blockDim.x + threadIdx.x;
    if (e >= NE) return;
    int s = ei[e], d = ei[NE + e];
    const float* qp = q + (size_t)d * 32;
    const float* kp = k + (size_t)s * 32;
    float acc = 0.f;
#pragma unroll
    for (int i = 0; i < 32; ++i) acc = fmaf(qp[i], kp[i], acc);
    acc *= 0.17677669529663689f;  // 1/sqrt(32)
    logit[e] = acc;
    atomicMax(&mbuf[d], fenc(acc));
}
__global__ void k_exp(const float* __restrict__ logit, const unsigned* __restrict__ mbuf,
                      const int* __restrict__ ei, float* __restrict__ expv,
                      float* __restrict__ sbuf) {
    int e = blockIdx.x * blockDim.x + threadIdx.x;
    if (e >= NE) return;
    int d = ei[NE + e];
    float ev = expf(logit[e] - fdec(mbuf[d]));
    expv[e] = ev;
    atomicAdd(&sbuf[d], ev);
}
__global__ void k_alpha(const float* __restrict__ expv, const float* __restrict__ sbuf,
                        const int* __restrict__ ei, float* __restrict__ alpha) {
    int e = blockIdx.x * blockDim.x + threadIdx.x;
    if (e >= NE) return;
    alpha[e] = expv[e] / (sbuf[ei[NE + e]] + 1e-16f);
}

// ---------------- h_mid = lrelu(LN16(ea@Wm1+bm1)) per edge ----------------
__global__ void __launch_bounds__(256) k_hmid(const float* __restrict__ ea,
    const float* __restrict__ Wm1, const float* __restrict__ bm1,
    const float* __restrict__ lnw, const float* __restrict__ lnb,
    float* __restrict__ hm) {
    __shared__ float sW[16 * 16];
    __shared__ float sb1[16], sw[16], sbb[16];
    for (int i = threadIdx.x; i < 256; i += 256) sW[i] = Wm1[i];
    if (threadIdx.x < 16) {
        sb1[threadIdx.x] = bm1[threadIdx.x];
        sw[threadIdx.x] = lnw[threadIdx.x];
        sbb[threadIdx.x] = lnb[threadIdx.x];
    }
    __syncthreads();
    int e = blockIdx.x * 256 + threadIdx.x;
    if (e >= NE) return;
    const float* r = ea + (size_t)e * 16;
    float acc[16];
#pragma unroll
    for (int o = 0; o < 16; ++o) acc[o] = sb1[o];
#pragma unroll
    for (int c = 0; c < 16; ++c) {
        float x = r[c];
#pragma unroll
        for (int o = 0; o < 16; ++o) acc[o] = fmaf(x, sW[c * 16 + o], acc[o]);
    }
    float m = 0.f;
#pragma unroll
    for (int o = 0; o < 16; ++o) m += acc[o];
    m *= (1.f / 16.f);
    float var = 0.f;
#pragma unroll
    for (int o = 0; o < 16; ++o) { float d = acc[o] - m; var = fmaf(d, d, var); }
    var *= (1.f / 16.f);
    float inv = rsqrtf(var + EPSV);
    float* op = hm + (size_t)e * 16;
#pragma unroll
    for (int o = 0; o < 16; ++o) op[o] = lrelu((acc[o] - m) * inv * sw[o] + sbb[o]);
}

// ---------------- f32 -> f16 node features ----------------
__global__ void k_cvt16(const float* __restrict__ h, _Float16* __restrict__ h16, int n) {
    int i = blockIdx.x * blockDim.x + threadIdx.x;
    if (i < n) h16[i] = (_Float16)h[i];
}

// ---------------- pack Wm2/bm2 into WMMA B-fragments (f16) ----------------
// bpack[((t*2+ot)*32+lane)*16 + i] ; K=c index: lane<16 -> c=i, lane>=16 -> c=16+i ; N=o=ot*16+(lane&15)
__global__ void k_packB(const float* __restrict__ Wm2, const float* __restrict__ bm2,
                        _Float16* __restrict__ bpack) {
    int tid = blockIdx.x * 64 + threadIdx.x;
    if (tid >= 17 * 2 * 32) return;
    int lane = tid & 31;
    int ot = (tid >> 5) & 1;
    int t = tid >> 6;
    int o = ot * 16 + (lane & 15);
    int cbase = (lane & 16) ? 16 : 0;
    _Float16* dst = bpack + (size_t)tid * 16;
#pragma unroll
    for (int i = 0; i < 16; ++i) {
        int c = cbase + i;
        float v = (t < 16) ? Wm2[t * 1024 + c * 32 + o] : bm2[c * 32 + o];
        dst[i] = (_Float16)v;
    }
}

// ---------------- WMMA message kernel: msg = (U @ Wm2r) * alpha scattered to dst ----------------
__global__ void __launch_bounds__(256) k_msg(const _Float16* __restrict__ h16,
                                             const float* __restrict__ hm,
                                             const _Float16* __restrict__ bpack,
                                             const float* __restrict__ alpha,
                                             const int* __restrict__ ei,
                                             float* __restrict__ newh) {
    __shared__ unsigned s_b[17 * 2 * 32 * 8];  // 34816 B of packed B fragments
    const unsigned* gb = (const unsigned*)bpack;
    for (int i = threadIdx.x; i < 17 * 2 * 32 * 8; i += 256) s_b[i] = gb[i];
    __syncthreads();

    const int lane = threadIdx.x & 31;
    const int wid = threadIdx.x >> 5;
    const int tile = blockIdx.x * 8 + wid;   // 16 edges per wave
    const int e0 = tile * 16;
    const int M = lane & 15;
    const bool hi = lane >= 16;
    const int e = e0 + M;
    const int s = ei[e];  // src node

    // A fragment: x_src in f16 (lane<16: K 0..15 ; lane>=16: K 16..31)
    v16h xa;
    {
        const _Float16* hp = h16 + (size_t)s * 32;
        ((uint4*)&xa)[0] = *(const uint4*)(hp + (hi ? 16 : 0));
        ((uint4*)&xa)[1] = *(const uint4*)(hp + (hi ? 24 : 8));
    }

    // per-edge h_mid row (16 values) + implicit 1.0 for the bias chunk
    _Float16 hmv[17];
    {
        const float* hp = hm + (size_t)e * 16;
#pragma unroll
        for (int t = 0; t < 16; ++t) hmv[t] = (_Float16)hp[t];
        hmv[16] = (_Float16)1.0f;
    }

    v8f acc0 = {0.f, 0.f, 0.f, 0.f, 0.f, 0.f, 0.f, 0.f};
    v8f acc1 = acc0;
    const _Float16* lb = (const _Float16*)s_b;

    // Double-buffered A-chunks: the pk_mul for chunk t+1 writes a different
    // VGPR set than the one the in-flight WMMAs of chunk t are reading,
    // eliminating the 4-NOP WMMA->VALU WAR hazard on the A registers.
    v16h acbuf[2];
    acbuf[0] = xa * hmv[0];
#pragma unroll
    for (int t = 0; t < 17; ++t) {
        v16h cur = acbuf[t & 1];
        if (t < 16) acbuf[(t + 1) & 1] = xa * hmv[t + 1];
        v16h b0, b1;
        const uint4* bp0 = (const uint4*)(lb + ((t * 2 + 0) * 32 + lane) * 16);
        const uint4* bp1 = (const uint4*)(lb + ((t * 2 + 1) * 32 + lane) * 16);
        ((uint4*)&b0)[0] = bp0[0]; ((uint4*)&b0)[1] = bp0[1];
        ((uint4*)&b1)[0] = bp1[0]; ((uint4*)&b1)[1] = bp1[1];
        acc0 = __builtin_amdgcn_wmma_f32_16x16x32_f16(false, cur, false, b0, (short)0, acc0, false, false);
        acc1 = __builtin_amdgcn_wmma_f32_16x16x32_f16(false, cur, false, b1, (short)0, acc1, false, false);
    }

    // D layout: VGPR v -> M = v + (hi?8:0), N = lane&15 ; scale by alpha, scatter-add to dst
    const int baseM = hi ? 8 : 0;
    const int n0 = lane & 15;
#pragma unroll
    for (int v = 0; v < 8; ++v) {
        const int me = e0 + baseM + v;
        const float al = alpha[me];
        const int d = ei[NE + me];
        atomicAdd(&newh[(size_t)d * 32 + n0], acc0[v] * al);
        atomicAdd(&newh[(size_t)d * 32 + 16 + n0], acc1[v] * al);
    }
}

// ---------------- pooling + head ----------------
__global__ void k_pool(const float* __restrict__ h, const int* __restrict__ batch,
                       float* __restrict__ gpool) {
    int t = blockIdx.x * blockDim.x + threadIdx.x;
    int node = t >> 5, f = t & 31;
    if (node >= NN) return;
    atomicAdd(&gpool[batch[node] * 32 + f], h[(size_t)node * 32 + f]);
}
__global__ void k_head(const float* __restrict__ gpool, const float* __restrict__ cnt,
                       const float* __restrict__ Wg1, const float* __restrict__ bg1,
                       const float* __restrict__ gw, const float* __restrict__ gb,
                       const float* __restrict__ Wg2, const float* __restrict__ bg2,
                       float* __restrict__ out) {
    int g = threadIdx.x;
    if (g >= NG) return;
    float c = fmaxf(cnt[g], 1.f);
    float xg[32];
#pragma unroll
    for (int f = 0; f < 32; ++f) xg[f] = gpool[g * 32 + f] / c;
    float v[32];
#pragma unroll
    for (int o = 0; o < 32; ++o) {
        float s = bg1[o];
#pragma unroll
        for (int f = 0; f < 32; ++f) s = fmaf(xg[f], Wg1[f * 32 + o], s);
        v[o] = s;
    }
    float m = 0.f;
#pragma unroll
    for (int o = 0; o < 32; ++o) m += v[o];
    m *= (1.f / 32.f);
    float var = 0.f;
#pragma unroll
    for (int o = 0; o < 32; ++o) { float d = v[o] - m; var = fmaf(d, d, var); }
    var *= (1.f / 32.f);
    float inv = rsqrtf(var + EPSV);
    float sacc = bg2[0];
#pragma unroll
    for (int o = 0; o < 32; ++o) {
        float y = lrelu((v[o] - m) * inv * gw[o] + gb[o]);
        sacc = fmaf(y, Wg2[o], sacc);
    }
    out[g] = sacc;
}

extern "C" void kernel_launch(void* const* d_in, const int* in_sizes, int n_in,
                              void* d_out, int out_size, void* d_ws, size_t ws_size,
                              hipStream_t stream) {
    (void)in_sizes; (void)n_in; (void)out_size; (void)ws_size;

    // ---- input unpack (setup_inputs()/make_params() insertion order) ----
    const float* x = (const float*)d_in[0];
    const float* eattr = (const float*)d_in[1];
    const float* Wn1 = (const float*)d_in[2];
    const float* bn1 = (const float*)d_in[3];
    const float* nnw = (const float*)d_in[4];
    const float* nnb = (const float*)d_in[5];
    const float* Wn2 = (const float*)d_in[6];
    const float* bn2 = (const float*)d_in[7];
    const float* We1 = (const float*)d_in[8];
    const float* be1 = (const float*)d_in[9];
    const float* enw = (const float*)d_in[10];
    const float* enb = (const float*)d_in[11];
    const float* We2 = (const float*)d_in[12];
    const float* be2 = (const float*)d_in[13];
    const float* cp[3][14];
    for (int L = 0; L < 3; ++L)
        for (int j = 0; j < 14; ++j) cp[L][j] = (const float*)d_in[14 + 14 * L + j];
    const float* Wg1 = (const float*)d_in[56];
    const float* bg1 = (const float*)d_in[57];
    const float* gw = (const float*)d_in[58];
    const float* gb = (const float*)d_in[59];
    const float* Wg2 = (const float*)d_in[60];
    const float* bg2 = (const float*)d_in[61];
    const int* ei = (const int*)d_in[62];
    const int* batch = (const int*)d_in[63];
    float* out = (float*)d_out;

    // ---- workspace carve ----
    char* w = (char*)d_ws;
    size_t off = 0;
    auto alloc = [&](size_t bytes) -> void* {
        void* p = w + off;
        off = (off + bytes + 255) & ~(size_t)255;
        return p;
    };
    float* h = (float*)alloc(NN * 32 * 4);
    float* h1 = (float*)alloc(NN * 32 * 4);
    float* q = (float*)alloc(NN * 32 * 4);
    float* kk = (float*)alloc(NN * 32 * 4);
    float* newh = (float*)alloc(NN * 32 * 4);
    _Float16* h16 = (_Float16*)alloc(NN * 32 * 2);
    float* ea = (float*)alloc((size_t)NE * 16 * 4);
    float* hm = (float*)alloc((size_t)NE * 16 * 4);
    float* logit = (float*)alloc(NE * 4);
    float* expv = (float*)alloc(NE * 4);
    float* alpha = (float*)alloc(NE * 4);
    unsigned* mbuf = (unsigned*)alloc(NN * 4);
    float* sbuf = (float*)alloc(NN * 4);
    float* cnt = (float*)alloc(NG * 4);
    float* gstat = (float*)alloc(2 * NG * 4);  // gsum | gsq
    float* gsum = gstat, *gsq = gstat + NG;
    float* gmi = (float*)alloc(2 * NG * 4);    // gmean | ginv
    float* gmean = gmi, *ginv = gmi + NG;
    float* gpool = (float*)alloc(NG * 32 * 4);
    _Float16* bpack = (_Float16*)alloc(17 * 2 * 32 * 16 * 2);

    const int TPB = 256;
    const int gNode32 = (NN * 32) / TPB;  // 2048
    const int gNode = NN / TPB;           // 64
    const int gEdge = NE / TPB;           // 512

    // ---- node counts ----
    k_fill_f32<<<1, 64, 0, stream>>>(cnt, 0.f, NG);
    k_count<<<gNode, TPB, 0, stream>>>(batch, cnt);

    // ---- node encoder ----
    k_lin<<<gNode32, TPB, 0, stream>>>(x, Wn1, bn1, h1, NN, 19);
    k_fill_f32<<<1, 128, 0, stream>>>(gstat, 0.f, 2 * NG);
    k_gln_reduce<<<gNode, TPB, 0, stream>>>(h1, batch, gsum, gsq);
    k_gln_final<<<1, 64, 0, stream>>>(gsum, gsq, cnt, gmean, ginv);
    k_gln_apply<<<gNode32, TPB, 0, stream>>>(h1, h1, batch, gmean, ginv, nnw, nnb, 1);
    k_lin<<<gNode32, TPB, 0, stream>>>(h1, Wn2, bn2, h, NN, 32);

    // ---- edge features (computed once, reused by all convs) ----
    k_edge_feat<<<gEdge, TPB, 0, stream>>>(h, ei, eattr, We1, be1, enw, enb, We2, be2, ea);

    // ---- 3 x EVTConv ----
    for (int L = 0; L < 3; ++L) {
        const float *Wq = cp[L][0], *bq = cp[L][1], *Wk = cp[L][2], *bk = cp[L][3];
        const float *Ws = cp[L][4], *bs = cp[L][5], *Wm1 = cp[L][6], *bm1 = cp[L][7];
        const float *lnw = cp[L][8], *lnb = cp[L][9], *Wm2 = cp[L][10], *bm2 = cp[L][11];
        const float *cnw = cp[L][12], *cnb = cp[L][13];

        k_lin<<<gNode32, TPB, 0, stream>>>(h, Wq, bq, q, NN, 32);
        k_lin<<<gNode32, TPB, 0, stream>>>(h, Wk, bk, kk, NN, 32);

        k_fill_u32<<<gNode, TPB, 0, stream>>>(mbuf, 0u, NN);
        k_fill_f32<<<gNode, TPB, 0, stream>>>(sbuf, 0.f, NN);
        k_logits<<<gEdge, TPB, 0, stream>>>(q, kk, ei, logit, mbuf);
        k_exp<<<gEdge, TPB, 0, stream>>>(logit, mbuf, ei, expv, sbuf);
        k_alpha<<<gEdge, TPB, 0, stream>>>(expv, sbuf, ei, alpha);

        k_hmid<<<gEdge, TPB, 0, stream>>>(ea, Wm1, bm1, lnw, lnb, hm);
        k_cvt16<<<gNode32, TPB, 0, stream>>>(h, h16, NN * 32);
        k_packB<<<17, 64, 0, stream>>>(Wm2, bm2, bpack);

        k_lin<<<gNode32, TPB, 0, stream>>>(h, Ws, bs, newh, NN, 32);  // root term
        k_msg<<<NE / 128, TPB, 0, stream>>>(h16, hm, bpack, alpha, ei, newh);

        k_fill_f32<<<1, 128, 0, stream>>>(gstat, 0.f, 2 * NG);
        k_gln_reduce<<<gNode, TPB, 0, stream>>>(newh, batch, gsum, gsq);
        k_gln_final<<<1, 64, 0, stream>>>(gsum, gsq, cnt, gmean, ginv);
        k_gln_apply<<<gNode32, TPB, 0, stream>>>(newh, h, batch, gmean, ginv, cnw, cnb,
                                                 (L < 2) ? 1 : 0);
    }

    // ---- pooling + head ----
    k_fill_f32<<<(NG * 32) / 64, 64, 0, stream>>>(gpool, 0.f, NG * 32);
    k_pool<<<gNode32, TPB, 0, stream>>>(h, batch, gpool);
    k_head<<<1, 64, 0, stream>>>(gpool, cnt, Wg1, bg1, gw, gb, Wg2, bg2, out);
}
